// StackDecoderLayer_35828617183552
// MI455X (gfx1250) — matrix operand order repaired and checked by reading, not compiled, not run on hardware
//
#include <hip/hip_runtime.h>
#include <hip/hip_bf16.h>

typedef __attribute__((ext_vector_type(16))) _Float16 v16h;
typedef __attribute__((ext_vector_type(8)))  float    v8f;

#define LDSP 40   // padded LDS row stride in halves (80B rows -> conflict-free, 16B aligned chunks)

// ---------------------------------------------------------------------------
// Generic batched WMMA GEMM:  C[z] = epilogue( A[z] (MxK) * B[z] (KxN) )
//   A: row-major, k contiguous, row stride lda
//   B: accessed as BT[n*ldbt + k*ldbk]  (ldbk==1 -> contiguous fast path)
//   batch z decomposed as (zb = z/BH, zh = z%BH); base offsets off = zb*X0 + zh*X1
// Tile: BM=128, BN=128, BK=32; 256 threads = 8 waves arranged 4x2.
// Wave (wm,wn) owns a 32x64 strip: 2 A-frags x 4 B-frags = 8 WMMA / K-step.
// LDS double-buffered: global loads for tile k+1 overlap WMMA on tile k.
// ---------------------------------------------------------------------------
struct GP {
  const _Float16* A; long lda; long aB0, aB1;
  const _Float16* Bt; long ldbt; long ldbk; long bB0, bB1;
  const float* bias;
  const float* resid; long ldres;
  float* outF; long ldf; long fB0, fB1;
  _Float16* outH; long ldh; long hB0, hB1;
  int M, N, K;
  float scale;
  int combine;   // 0: resid + v   1: max(resid, v)
  int act;       // 0: none  1: relu
  int causal;    // scores epilogue: NEG above diagonal + key mask
  const unsigned char* kmask; int BH; int Skm;
};

__device__ __forceinline__ v16h frag_a(const _Float16* As, int m0, int lane) {
  // 16-bit A 16x32 layout: lane l holds row m0+(l&15); halves 0-7 = K kb..kb+7,
  // halves 8-15 = K kb+16..kb+23, kb = (l>>4)*8
  int r  = m0 + (lane & 15);
  int kb = (lane >> 4) * 8;
  const _Float16* p = As + r * LDSP + kb;
  v16h f;
  ((uint4*)&f)[0] = *(const uint4*)p;
  ((uint4*)&f)[1] = *(const uint4*)(p + 16);
  return f;
}
__device__ __forceinline__ v16h frag_b(const _Float16* Bs, int n0, int lane) {
  // 16-bit B 32x16 layout: lane l holds col n0+(l&15); halves 0-7 = K kb..kb+7,
  // halves 8-15 = K kb+16..kb+23, kb = (l>>4)*8
  int c  = n0 + (lane & 15);
  int kb = (lane >> 4) * 8;
  const _Float16* p = Bs + c * LDSP + kb;
  v16h f;
  ((uint4*)&f)[0] = *(const uint4*)p;
  ((uint4*)&f)[1] = *(const uint4*)(p + 16);
  return f;
}

__global__ __launch_bounds__(256) void gemm_wmma(GP p) {
  __shared__ _Float16 As[2][128 * LDSP];
  __shared__ _Float16 Bs[2][128 * LDSP];

  const int tid  = threadIdx.x;
  const int wave = tid >> 5;
  const int lane = tid & 31;
  const int wm   = wave >> 1;          // 0..3  -> M strip of 32
  const int wn   = wave & 1;           // 0..1  -> N strip of 64
  const int m0   = blockIdx.y * 128;
  const int n0   = blockIdx.x * 128;
  const int z    = blockIdx.z;
  const int zb   = z / p.BH;
  const int zh   = z % p.BH;

  const _Float16* A  = p.A  + (long)zb * p.aB0 + (long)zh * p.aB1;
  const _Float16* Bt = p.Bt + (long)zb * p.bB0 + (long)zh * p.bB1;

  v8f acc[8] = {};

  const int ra = tid >> 1, ka = (tid & 1) * 16;   // A: 16 halves / thread (128 rows)
  const int nb = tid >> 1, kb = (tid & 1) * 16;   // B: 16 halves / thread (128 n-rows)
  const uint4 zero4 = {0u, 0u, 0u, 0u};

  uint4 a0, a1, b0, b1;
  const int nk = p.K >> 5;

  // ---- stage loader: global -> registers for K-tile kt ----
  auto load_regs = [&](int kt) {
    int k0 = kt << 5;
    const _Float16* ga = A + (long)(m0 + ra) * p.lda + k0 + ka;
    a0 = *(const uint4*)ga;
    a1 = *(const uint4*)(ga + 8);
    if (kt + 1 < nk) __builtin_prefetch(ga + 32, 0, 1);     // global_prefetch_b8
    if (n0 + nb < p.N) {
      if (p.ldbk == 1) {
        const _Float16* gb = Bt + (long)(n0 + nb) * p.ldbt + k0 + kb;
        b0 = *(const uint4*)gb;
        b1 = *(const uint4*)(gb + 8);
        if (kt + 1 < nk) __builtin_prefetch(gb + 32, 0, 1);
      } else {
        _Float16 tmp[16];
        #pragma unroll
        for (int i = 0; i < 16; ++i)
          tmp[i] = Bt[(long)(n0 + nb) * p.ldbt + (long)(k0 + kb + i) * p.ldbk];
        b0 = *(const uint4*)&tmp[0];
        b1 = *(const uint4*)&tmp[8];
      }
    } else { b0 = zero4; b1 = zero4; }
  };
  // ---- stage store: registers -> LDS buffer s ----
  auto store_lds = [&](int s) {
    *(uint4*)&As[s][ra * LDSP + ka]     = a0;
    *(uint4*)&As[s][ra * LDSP + ka + 8] = a1;
    *(uint4*)&Bs[s][nb * LDSP + kb]     = b0;
    *(uint4*)&Bs[s][nb * LDSP + kb + 8] = b1;
  };

  load_regs(0);
  store_lds(0);
  __syncthreads();

  for (int kt = 0; kt < nk; ++kt) {
    const int cur = kt & 1;
    if (kt + 1 < nk) load_regs(kt + 1);          // overlap with WMMA below

    const _Float16* Ab = As[cur];
    const _Float16* Bb = Bs[cur];
    v16h fa0 = frag_a(Ab, wm * 32,      lane);
    v16h fa1 = frag_a(Ab, wm * 32 + 16, lane);
    #pragma unroll
    for (int j = 0; j < 4; ++j) {
      v16h fb = frag_b(Bb, wn * 64 + j * 16, lane);
      acc[j]     = __builtin_amdgcn_wmma_f32_16x16x32_f16(
          false, fa0, false, fb, (short)0, acc[j],     false, false);
      acc[4 + j] = __builtin_amdgcn_wmma_f32_16x16x32_f16(
          false, fa1, false, fb, (short)0, acc[4 + j], false, false);
    }
    if (kt + 1 < nk) store_lds((kt + 1) & 1);
    __syncthreads();
  }

  // --- epilogue (C layout: VGPR i -> row i + (lane>>4)*8, col = lane&15) ---
  const int rl = (lane >> 4) * 8;
  const int cl = lane & 15;
  const long fOff = (long)zb * p.fB0 + (long)zh * p.fB1;
  const long hOff = (long)zb * p.hB0 + (long)zh * p.hB1;
  #pragma unroll
  for (int j = 0; j < 4; ++j) {
    int col = n0 + wn * 64 + j * 16 + cl;
    if (col >= p.N) continue;
    float bv = p.bias ? p.bias[col] : 0.0f;
    #pragma unroll
    for (int h = 0; h < 2; ++h) {
      #pragma unroll
      for (int i = 0; i < 8; ++i) {
        int row = m0 + wm * 32 + h * 16 + i + rl;
        float v = acc[h * 4 + j][i] * p.scale + bv;
        if (p.causal) {
          if (col > row) v = -1.0e9f;
          if (p.kmask && p.kmask[(long)zb * p.Skm + col]) v = -1.0e9f;
        }
        if (p.resid) {
          float r = p.resid[(long)row * p.ldres + col];
          v = p.combine ? fmaxf(r, v) : (r + v);
        }
        if (p.act == 1) v = fmaxf(v, 0.0f);
        if (p.outF) p.outF[fOff + (long)row * p.ldf + col] = v;
        if (p.outH) p.outH[hOff + (long)row * p.ldh + col] = (_Float16)v;
      }
    }
  }
}

// ---------------------------------------------------------------------------
// Elementwise / reduction kernels
// ---------------------------------------------------------------------------
__global__ __launch_bounds__(256) void ln_kernel(const float* __restrict__ x,
                                                 const float* __restrict__ g,
                                                 const float* __restrict__ b,
                                                 _Float16* __restrict__ out) {
  const int C = 512;
  long r = blockIdx.x; int tid = threadIdx.x;
  __shared__ float s1[256], s2[256];
  float v0 = x[r * C + tid], v1 = x[r * C + tid + 256];
  s1[tid] = v0 + v1;
  s2[tid] = v0 * v0 + v1 * v1;
  __syncthreads();
  for (int o = 128; o > 0; o >>= 1) {
    if (tid < o) { s1[tid] += s1[tid + o]; s2[tid] += s2[tid + o]; }
    __syncthreads();
  }
  float m   = s1[0] * (1.0f / C);
  float var = s2[0] * (1.0f / C) - m * m;
  float inv = rsqrtf(var + 1e-5f);
  out[r * C + tid]       = (_Float16)((v0 - m) * inv * g[tid] + b[tid]);
  out[r * C + tid + 256] = (_Float16)((v1 - m) * inv * g[tid + 256] + b[tid + 256]);
}

__global__ void softmax_kernel(const float* __restrict__ in, _Float16* __restrict__ out, int N) {
  extern __shared__ float red[];
  long r = blockIdx.x; int tid = threadIdx.x;
  float v = in[r * N + tid];
  red[tid] = v; __syncthreads();
  for (int o = N >> 1; o > 0; o >>= 1) {
    if (tid < o) red[tid] = fmaxf(red[tid], red[tid + o]);
    __syncthreads();
  }
  float mx = red[0]; __syncthreads();
  float e = __expf(v - mx);
  red[tid] = e; __syncthreads();
  for (int o = N >> 1; o > 0; o >>= 1) {
    if (tid < o) red[tid] += red[tid + o];
    __syncthreads();
  }
  out[r * N + tid] = (_Float16)(e / red[0]);
}

__global__ void nonsat_kernel(const float* __restrict__ in, float* __restrict__ outF,
                              _Float16* h1, long ldh1, long off1,
                              _Float16* h2, long ldh2, long total, int C) {
  long id = (long)blockIdx.x * 256 + threadIdx.x;
  if (id >= total) return;
  long r = id / C; int c = (int)(id % C);
  float xv = in[id];
  float y = xv;
  #pragma unroll
  for (int i = 0; i < 16; ++i) {
    float y2 = y * y;
    y = (0.66666667f * y * y2 + xv) / (y2 + 1.0f);
  }
  if (outF) outF[id] = y;
  if (h1) h1[r * ldh1 + off1 + c] = (_Float16)y;
  if (h2) h2[r * ldh2 + c] = (_Float16)y;
}

__global__ void ctrl_kernel(const float* __restrict__ hidden, const float* __restrict__ Aw,
                            const float* __restrict__ Ab, float* __restrict__ ctrl, int R) {
  int r = blockIdx.x * 256 + threadIdx.x;
  if (r >= R) return;
  float a0 = Ab[0], a1 = Ab[1], a2 = Ab[2];
  const float* h = hidden + (long)r * 512;
  for (int k = 0; k < 512; ++k) {
    float hv = h[k];
    a0 += hv * Aw[k * 3 + 0];
    a1 += hv * Aw[k * 3 + 1];
    a2 += hv * Aw[k * 3 + 2];
  }
  float mx = fmaxf(a0, fmaxf(a1, a2));
  float e0 = __expf(a0 - mx), e1 = __expf(a1 - mx), e2 = __expf(a2 - mx);
  float inv = 1.0f / (e0 + e1 + e2);
  ctrl[(long)r * 3 + 0] = e0 * inv;
  ctrl[(long)r * 3 + 1] = e1 * inv;
  ctrl[(long)r * 3 + 2] = e2 * inv;
}

__global__ void cvtT_kernel(const float* __restrict__ w, _Float16* __restrict__ wt,
                            int K, int N) {
  long id = (long)blockIdx.x * 256 + threadIdx.x;
  if (id >= (long)K * N) return;
  long n = id / K, k = id % K;
  wt[id] = (_Float16)w[k * N + n];
}

__global__ void stackx_kernel(const float* __restrict__ stack_prev, _Float16* __restrict__ sx,
                              int S, int B) {
  long id = (long)blockIdx.x * 256 + threadIdx.x;
  if (id >= (long)S * B * 64) return;
  int w = (int)(id & 63);
  long r = id >> 6;                 // r = s*B + b
  long s = r / B, b = r % B;
  sx[id] = (_Float16)stack_prev[(b * S + s) * 2048 + w];   // depth slot 0
}

// stack = c2*prev + c0*up + c1*down ; single pass over prev via LDS staging
__global__ __launch_bounds__(256) void stack_kernel(const float* __restrict__ prev,
                                                    const float* __restrict__ sinp,
                                                    const float* __restrict__ ctrl,
                                                    float* __restrict__ out, int S, int B) {
  __shared__ float pl[2048];
  __shared__ float sl[64];
  int bid = blockIdx.x;                 // bid = b*S + s
  int b = bid / S, s = bid % S;
  const float* pp = prev + (long)bid * 2048;
  for (int e = threadIdx.x; e < 2048; e += 256) pl[e] = pp[e];
  long rsb = (long)s * B + b;
  if (threadIdx.x < 64) sl[threadIdx.x] = sinp[rsb * 64 + threadIdx.x];
  __syncthreads();
  float c0 = ctrl[rsb * 3 + 0], c1 = ctrl[rsb * 3 + 1], c2 = ctrl[rsb * 3 + 2];
  for (int e = threadIdx.x; e < 2048; e += 256) {
    int d = e >> 6, w = e & 63;
    float up = (d == 0)  ? sl[w] : pl[e - 64];
    float dn = (d == 31) ? 0.0f  : pl[e + 64];
    out[(long)bid * 2048 + e] = c2 * pl[e] + c0 * up + c1 * dn;
  }
}

// ---------------------------------------------------------------------------
// Host orchestration
// ---------------------------------------------------------------------------
static void launch_gemm(const GP& g, int Z, hipStream_t s) {
  dim3 grid((g.N + 127) / 128, g.M / 128, Z);
  gemm_wmma<<<grid, 256, 0, s>>>(g);
}
static GP mk() { GP g{}; g.scale = 1.0f; g.BH = 1; g.Skm = 1; return g; }

extern "C" void kernel_launch(void* const* d_in, const int* in_sizes, int n_in,
                              void* d_out, int out_size, void* d_ws, size_t ws_size,
                              hipStream_t stream) {
  const int S = 256, B = 64, E = 512, H = 8, R = S * B;
  const int DH = 512, SW = 64, DSS = 128, DFF = 2048;

  const float* x_in       = (const float*)d_in[0];
  const float* stack_prev = (const float*)d_in[1];
  const unsigned char* km = (const unsigned char*)d_in[2];
  const float* ln1_g = (const float*)d_in[3];
  const float* ln1_b = (const float*)d_in[4];
  const float* in_proj_w = (const float*)d_in[5];
  const float* in_proj_b = (const float*)d_in[6];
  const float* out_w = (const float*)d_in[7];
  const float* out_b = (const float*)d_in[8];
  const float* W_w = (const float*)d_in[9];
  const float* W_b = (const float*)d_in[10];
  const float* P_w = (const float*)d_in[11];
  const float* P_b = (const float*)d_in[12];
  const float* V_w = (const float*)d_in[13];
  const float* U_w = (const float*)d_in[14];
  const float* A_w = (const float*)d_in[15];
  const float* A_b = (const float*)d_in[16];
  const float* D_w = (const float*)d_in[17];
  const float* D_b = (const float*)d_in[18];
  const float* ln2_g = (const float*)d_in[19];
  const float* ln2_b = (const float*)d_in[20];
  const float* ff1_w = (const float*)d_in[21];
  const float* ff1_b = (const float*)d_in[22];
  const float* ff2_w = (const float*)d_in[23];
  const float* ff2_b = (const float*)d_in[24];

  float* out_x     = (float*)d_out;
  float* out_stack = (float*)d_out + (long)R * E;

  // ---- workspace bump allocator ----
  char* wp = (char*)d_ws;
  auto alloc = [&](size_t bytes) -> void* {
    void* p = (void*)wp;
    wp += (bytes + 255) & ~(size_t)255;
    return p;
  };
  _Float16* ipT   = (_Float16*)alloc((size_t)3 * E * E * 2);          // 1536x512
  _Float16* outT  = (_Float16*)alloc((size_t)E * E * 2);              // 512x512
  _Float16* WT    = (_Float16*)alloc((size_t)DH * E * 2);             // 512x512
  _Float16* PT    = (_Float16*)alloc((size_t)DH * SW * 2);            // 512x64
  _Float16* VT    = (_Float16*)alloc((size_t)DSS * (E + DH) * 2);     // 128x1024
  _Float16* UT    = (_Float16*)alloc((size_t)E * DSS * 2);            // 512x128
  _Float16* DT    = (_Float16*)alloc((size_t)SW * DH * 2);            // 64x512
  _Float16* ff1T  = (_Float16*)alloc((size_t)DFF * E * 2);            // 2048x512
  _Float16* ff2T  = (_Float16*)alloc((size_t)E * DFF * 2);            // 512x2048
  _Float16* xn_h  = (_Float16*)alloc((size_t)R * E * 2);
  _Float16* qkv_h = (_Float16*)alloc((size_t)R * 3 * E * 2);
  float*    scores= (float*)   alloc((size_t)B * H * S * S * 4);
  _Float16* probs = (_Float16*)alloc((size_t)B * H * S * S * 2);
  _Float16* ao_h  = (_Float16*)alloc((size_t)R * E * 2);
  float*    x     = (float*)   alloc((size_t)R * E * 4);
  _Float16* vx_h  = (_Float16*)alloc((size_t)R * (E + DH) * 2);
  float*    hpre  = (float*)   alloc((size_t)R * DH * 4);
  float*    hid   = (float*)   alloc((size_t)R * DH * 4);
  _Float16* hid_h = (_Float16*)alloc((size_t)R * DH * 2);
  float*    logit = (float*)   alloc((size_t)R * DSS * 4);
  _Float16* p128  = (_Float16*)alloc((size_t)R * DSS * 2);
  float*    spre  = (float*)   alloc((size_t)R * SW * 4);
  float*    sinp  = (float*)   alloc((size_t)R * SW * 4);
  float*    ctrl  = (float*)   alloc((size_t)R * 3 * 4);
  _Float16* h1_h  = (_Float16*)alloc((size_t)R * DFF * 2);
  _Float16* xn2_h = (_Float16*)alloc((size_t)R * E * 2);
  _Float16* sx_h  = (_Float16*)alloc((size_t)R * SW * 2);

  auto cvt = [&](const float* w, _Float16* wt, int K, int N) {
    long tot = (long)K * N;
    cvtT_kernel<<<(tot + 255) / 256, 256, 0, stream>>>(w, wt, K, N);
  };
  // ---- weight transpose + f16 convert ----
  cvt(in_proj_w, ipT, E, 3 * E);
  cvt(out_w, outT, E, E);
  cvt(W_w, WT, E, DH);
  cvt(P_w, PT, SW, DH);
  cvt(V_w, VT, E + DH, DSS);
  cvt(U_w, UT, DSS, E);
  cvt(D_w, DT, DH, SW);
  cvt(ff1_w, ff1T, E, DFF);
  cvt(ff2_w, ff2T, DFF, E);

  // ---- 1. LN1 ----
  ln_kernel<<<R, 256, 0, stream>>>(x_in, ln1_g, ln1_b, xn_h);

  // ---- 2. QKV GEMM ----
  { GP g = mk();
    g.A = xn_h; g.lda = E; g.Bt = ipT; g.ldbt = E; g.ldbk = 1;
    g.bias = in_proj_b; g.outH = qkv_h; g.ldh = 3 * E;
    g.M = R; g.N = 3 * E; g.K = E;
    launch_gemm(g, 1, stream); }

  // ---- 3. attention scores (batched over b,h; fused scale/causal/key-mask) ----
  { GP g = mk();
    g.A = qkv_h; g.lda = (long)B * 3 * E; g.aB0 = 3 * E; g.aB1 = 64;
    g.Bt = qkv_h + E; g.ldbt = (long)B * 3 * E; g.ldbk = 1; g.bB0 = 3 * E; g.bB1 = 64;
    g.outF = scores; g.ldf = S; g.fB0 = (long)H * S * S; g.fB1 = (long)S * S;
    g.M = S; g.N = S; g.K = 64;
    g.scale = 0.125f; g.causal = 1; g.kmask = km; g.BH = H; g.Skm = S;
    launch_gemm(g, B * H, stream); }

  // ---- 4. softmax over keys ----
  softmax_kernel<<<(long)B * H * S, S, S * sizeof(float), stream>>>(scores, probs, S);

  // ---- 5. attn @ V (B operand gathered: strided over t) ----
  { GP g = mk();
    g.A = probs; g.lda = S; g.aB0 = (long)H * S * S; g.aB1 = (long)S * S;
    g.Bt = qkv_h + 2 * E; g.ldbt = 1; g.ldbk = (long)B * 3 * E; g.bB0 = 3 * E; g.bB1 = 64;
    g.outH = ao_h; g.ldh = (long)B * E; g.hB0 = E; g.hB1 = 64;
    g.M = S; g.N = 64; g.K = S; g.BH = H;
    launch_gemm(g, B * H, stream); }

  // ---- 6. out-proj + residual; also emit f16 x into vx[:,0:512] ----
  { GP g = mk();
    g.A = ao_h; g.lda = E; g.Bt = outT; g.ldbt = E; g.ldbk = 1;
    g.bias = out_b; g.resid = x_in; g.ldres = E;
    g.outF = x; g.ldf = E; g.outH = vx_h; g.ldh = E + DH;
    g.M = R; g.N = E; g.K = E;
    launch_gemm(g, 1, stream); }

  // ---- 7. stack top slice -> f16 (reordered to (s,b)) ----
  stackx_kernel<<<((long)R * SW + 255) / 256, 256, 0, stream>>>(stack_prev, sx_h, S, B);

  // ---- 8. hidden pre-act: x@W + b ; += stack_x@P + b ----
  { GP g = mk();
    g.A = vx_h; g.lda = E + DH; g.Bt = WT; g.ldbt = E; g.ldbk = 1;
    g.bias = W_b; g.outF = hpre; g.ldf = DH;
    g.M = R; g.N = DH; g.K = E;
    launch_gemm(g, 1, stream); }
  { GP g = mk();
    g.A = sx_h; g.lda = SW; g.Bt = PT; g.ldbt = SW; g.ldbk = 1;
    g.bias = P_b; g.resid = hpre; g.ldres = DH;
    g.outF = hpre; g.ldf = DH;
    g.M = R; g.N = DH; g.K = SW;
    launch_gemm(g, 1, stream); }

  // ---- 9. nonsat -> hidden (f32, f16, and vx[:,512:1024]) ----
  nonsat_kernel<<<((long)R * DH + 255) / 256, 256, 0, stream>>>(
      hpre, hid, vx_h, E + DH, E, hid_h, DH, (long)R * DH, DH);

  // ---- 10. V logits + softmax(128) + U mix, x = max(x, y) ----
  { GP g = mk();
    g.A = vx_h; g.lda = E + DH; g.Bt = VT; g.ldbt = E + DH; g.ldbk = 1;
    g.outF = logit; g.ldf = DSS;
    g.M = R; g.N = DSS; g.K = E + DH;
    launch_gemm(g, 1, stream); }
  softmax_kernel<<<R, DSS, DSS * sizeof(float), stream>>>(logit, p128, DSS);
  { GP g = mk();
    g.A = p128; g.lda = DSS; g.Bt = UT; g.ldbt = DSS; g.ldbk = 1;
    g.resid = x; g.ldres = E; g.combine = 1;
    g.outF = x; g.ldf = E;
    g.M = R; g.N = E; g.K = DSS;
    launch_gemm(g, 1, stream); }

  // ---- 11. stack input: nonsat(hidden@D + b) ----
  { GP g = mk();
    g.A = hid_h; g.lda = DH; g.Bt = DT; g.ldbt = DH; g.ldbk = 1;
    g.bias = D_b; g.outF = spre; g.ldf = SW;
    g.M = R; g.N = SW; g.K = DH;
    launch_gemm(g, 1, stream); }
  nonsat_kernel<<<((long)R * SW + 255) / 256, 256, 0, stream>>>(
      spre, sinp, (_Float16*)nullptr, 0, 0, (_Float16*)nullptr, 0, (long)R * SW, SW);

  // ---- 12. ctrl = softmax(hidden@A + b) ----
  ctrl_kernel<<<(R + 255) / 256, 256, 0, stream>>>(hid, A_w, A_b, ctrl, R);

  // ---- 13. stack shift/update (pure bandwidth, LDS-staged) ----
  stack_kernel<<<B * S, 256, 0, stream>>>(stack_prev, sinp, ctrl, out_stack, S, B);

  // ---- 14. LN2 + FFN ----
  ln_kernel<<<R, 256, 0, stream>>>(x, ln2_g, ln2_b, xn2_h);
  { GP g = mk();
    g.A = xn2_h; g.lda = E; g.Bt = ff1T; g.ldbt = E; g.ldbk = 1;
    g.bias = ff1_b; g.act = 1; g.outH = h1_h; g.ldh = DFF;
    g.M = R; g.N = DFF; g.K = E;
    launch_gemm(g, 1, stream); }
  { GP g = mk();
    g.A = h1_h; g.lda = DFF; g.Bt = ff2T; g.ldbt = DFF; g.ldbk = 1;
    g.bias = ff2_b; g.resid = x; g.ldres = E;
    g.outF = out_x; g.ldf = E;
    g.M = R; g.N = E; g.K = DFF;
    launch_gemm(g, 1, stream); }
  (void)in_sizes; (void)n_in; (void)out_size; (void)ws_size;
}
